// FinGAT_14551349198957
// MI455X (gfx1250) — matrix operand order repaired and checked by reading, not compiled, not run on hardware
//
#include <hip/hip_runtime.h>

// ---------------------------------------------------------------------------
// FinGAT on MI455X (gfx1250, wave32).  All GEMMs -> v_wmma_f32_16x16x32_f16.
// f16 operands / f32 accumulate; weights pre-cast+transposed into ws once.
// GRU recurrences keep h and the [Wih;Whh] operand LDS-resident per WGP.
// ---------------------------------------------------------------------------

typedef __attribute__((ext_vector_type(16))) _Float16 v16h;
typedef __attribute__((ext_vector_type(8)))  float    v8f;

#define DEVINL __device__ __forceinline__

DEVINL float sigmoidf_(float x) { return 1.f / (1.f + __expf(-x)); }

DEVINL v8f wmma16(v16h a, v16h b, v8f c) {
  return __builtin_amdgcn_wmma_f32_16x16x32_f16(false, a, false, b, (short)0, c,
                                                false, false);
}

// load 16 halfs as two 16B chunks (p0: elems0-7, p1: elems8-15)
DEVINL v16h ldpair(const _Float16* p0, const _Float16* p1) {
  v16h r;
  ((uint4*)&r)[0] = ((const uint4*)p0)[0];
  ((uint4*)&r)[1] = ((const uint4*)p1)[0];
  return r;
}
// A fragment (16xK row-major src, ld halfs): ISA 16-bit A layout
DEVINL v16h loadA(const _Float16* base, int ld, int m, int half, int c) {
  const _Float16* p = base + (size_t)m * ld + c * 32 + half * 8;
  return ldpair(p, p + 16);
}
// B fragment (KxN row-major src, ld halfs): lane%16 = N col, 16 K in table order
DEVINL v16h loadB(const _Float16* base, int ld, int lane, int c, int nt) {
  v16h r;
  const int n = nt * 16 + (lane & 15), h = lane >> 4;
  #pragma unroll
  for (int e = 0; e < 16; ++e) {
    int k = c * 32 + (e < 8 ? h * 8 + e : 16 + h * 8 + (e - 8));
    r[e] = base[(size_t)k * ld + n];
  }
  return r;
}

// ------------------------------ prep kernels -------------------------------
__global__ void k_cast16(const float* __restrict__ src, _Float16* __restrict__ dst, int n) {
  int i = blockIdx.x * 256 + threadIdx.x;
  if (i < n) dst[i] = (_Float16)src[i];
}
// dst[k*N+n] = src[n*K+k]   (dst: K rows x N cols f16; src: N rows x K cols f32)
__global__ void k_tcast(_Float16* __restrict__ dst, const float* __restrict__ src, int K, int N) {
  int i = blockIdx.x * 256 + threadIdx.x;
  if (i >= K * N) return;
  int k = i / N, n = i - k * N;
  dst[i] = (_Float16)src[n * K + k];
}
__global__ void k_zeroh(_Float16* __restrict__ dst, int n) {
  int i = blockIdx.x * 256 + threadIdx.x;
  if (i < n) dst[i] = (_Float16)0.f;
}

// ------------------- K1: per-window GRU (D=5) + attention ------------------
// 16 sequences/WG, 4 waves. A = [x_t(16) | pad(16) | h(128)] -> 5 K-chunks.
// B1 (160x384 f16) LDS resident.  n-gate split: chunk0 -> inn, chunks1-4 -> hn.
__global__ void __launch_bounds__(128)
k1_window_gru(const float* __restrict__ x, const _Float16* __restrict__ B1,
              const float* __restrict__ bih, const float* __restrict__ bhh,
              const float* __restrict__ aw, const float* __restrict__ ab,
              float* __restrict__ shortOut) {
  extern __shared__ char smem[];
  _Float16* lB = (_Float16*)smem;               // 160x384 = 122880 B
  _Float16* lA = (_Float16*)(smem + 122880);    // 16x160  = 5120 B
  float* lRZ = (float*)(smem + 128000);         // 16x256
  float* lNX = (float*)(smem + 144384);         // 16x128
  float* lNH = (float*)(smem + 152576);         // 16x128
  float* lH  = (float*)(smem + 160768);         // 5x16x128
  float* lSc = (float*)(smem + 201728);         // 80

  const int tid = threadIdx.x;
  const int seq0 = blockIdx.x * 16;

  { const uint4* s = (const uint4*)B1; uint4* d = (uint4*)lB;
    for (int i = tid; i < 7680; i += 128) d[i] = s[i]; }
  { uint4 z; z.x = z.y = z.z = z.w = 0u; uint4* d = (uint4*)lA;
    for (int i = tid; i < 320; i += 128) d[i] = z; }
  __syncthreads();

  const int w = __builtin_amdgcn_readfirstlane(tid >> 5);
  const int lane = tid & 31, mrow = lane & 15, half = lane >> 4;

  for (int t = 0; t < 5; ++t) {
    // stage x_t into A (K 0..15)
    #pragma unroll
    for (int u = 0; u < 2; ++u) {
      int idx = tid * 2 + u, m = idx >> 4, f = idx & 15;
      lA[m * 160 + f] = (_Float16)x[((size_t)(seq0 + m) * 5 + t) * 16 + f];
    }
    __syncthreads();

    v16h af[5];
    #pragma unroll
    for (int c = 0; c < 5; ++c) af[c] = loadA(lA, 160, mrow, half, c);
    // r/z tiles: nt = w*4 + i  (0..15), all 5 chunks
    #pragma unroll
    for (int i = 0; i < 4; ++i) {
      int nt = w * 4 + i;
      v8f acc = {};
      #pragma unroll
      for (int c = 0; c < 5; ++c) acc = wmma16(af[c], loadB(lB, 384, lane, c, nt), acc);
      #pragma unroll
      for (int r = 0; r < 8; ++r) lRZ[(r + 8 * half) * 256 + nt * 16 + mrow] = acc[r];
    }
    // n tiles: nt = 16 + w*2 + i, split accumulators
    #pragma unroll
    for (int i = 0; i < 2; ++i) {
      int nt = 16 + w * 2 + i;
      v8f ax = {}, ah = {};
      ax = wmma16(af[0], loadB(lB, 384, lane, 0, nt), ax);
      #pragma unroll
      for (int c = 1; c < 5; ++c) ah = wmma16(af[c], loadB(lB, 384, lane, c, nt), ah);
      #pragma unroll
      for (int r = 0; r < 8; ++r) {
        lNX[(r + 8 * half) * 128 + (nt - 16) * 16 + mrow] = ax[r];
        lNH[(r + 8 * half) * 128 + (nt - 16) * 16 + mrow] = ah[r];
      }
    }
    __syncthreads();
    // gates
    #pragma unroll
    for (int u = 0; u < 16; ++u) {
      int idx = tid * 16 + u, m = idx >> 7, j = idx & 127;
      float rv = sigmoidf_(lRZ[m * 256 + j] + bih[j] + bhh[j]);
      float zv = sigmoidf_(lRZ[m * 256 + 128 + j] + bih[128 + j] + bhh[128 + j]);
      float nv = tanhf(lNX[m * 128 + j] + bih[256 + j] + rv * (lNH[m * 128 + j] + bhh[256 + j]));
      float hp = (t == 0) ? 0.f : lH[((t - 1) * 16 + m) * 128 + j];
      float hn = (1.f - zv) * nv + zv * hp;
      lH[(t * 16 + m) * 128 + j] = hn;
      lA[m * 160 + 32 + j] = (_Float16)hn;
    }
    __syncthreads();
  }
  // temporal attention over 5 steps
  if (tid < 80) {
    int t = tid >> 4, m = tid & 15;
    float s = ab[0];
    const float* hp = &lH[(t * 16 + m) * 128];
    for (int j = 0; j < 128; ++j) s += hp[j] * aw[j];
    lSc[t * 16 + m] = s;
  }
  __syncthreads();
  if (tid < 16) {
    int m = tid;
    float mx = lSc[m];
    for (int t = 1; t < 5; ++t) mx = fmaxf(mx, lSc[t * 16 + m]);
    float e[5], den = 0.f;
    for (int t = 0; t < 5; ++t) { e[t] = __expf(lSc[t * 16 + m] - mx); den += e[t]; }
    float inv = 1.f / den;
    for (int t = 0; t < 5; ++t) lSc[t * 16 + m] = e[t] * inv;
  }
  __syncthreads();
  #pragma unroll
  for (int u = 0; u < 16; ++u) {
    int idx = tid * 16 + u, m = idx >> 7, j = idx & 127;
    float s = 0.f;
    #pragma unroll
    for (int t = 0; t < 5; ++t) s += lSc[t * 16 + m] * lH[(t * 16 + m) * 128 + j];
    shortOut[(size_t)(seq0 + m) * 128 + j] = s;
  }
}

// ---------------- K2a: Wh = last @ Wintra, plus s1/s2 dots -----------------
__global__ void __launch_bounds__(32)
k2a_wh(const float* __restrict__ shortF, const _Float16* __restrict__ W16,
       const float* __restrict__ avec, _Float16* __restrict__ Wh16,
       float* __restrict__ s1, float* __restrict__ s2) {
  __shared__ float lWh[16 * 128];
  const int lane = threadIdx.x, mrow = lane & 15, half = lane >> 4;
  const int i0 = blockIdx.x * 16;
  v16h af[4];
  {
    const float* rp = shortF + ((size_t)(i0 + mrow) * 32 + 31) * 128;
    #pragma unroll
    for (int c = 0; c < 4; ++c) {
      v16h a;
      #pragma unroll
      for (int e = 0; e < 8; ++e) a[e] = (_Float16)rp[c * 32 + half * 8 + e];
      #pragma unroll
      for (int e = 0; e < 8; ++e) a[8 + e] = (_Float16)rp[c * 32 + 16 + half * 8 + e];
      af[c] = a;
    }
  }
  #pragma unroll
  for (int nt = 0; nt < 8; ++nt) {
    v8f acc = {};
    #pragma unroll
    for (int c = 0; c < 4; ++c) acc = wmma16(af[c], loadB(W16, 128, lane, c, nt), acc);
    #pragma unroll
    for (int r = 0; r < 8; ++r) lWh[(r + 8 * half) * 128 + nt * 16 + mrow] = acc[r];
  }
  __syncthreads();
  for (int i = lane; i < 2048; i += 32)
    Wh16[((size_t)(i0 + (i >> 7))) * 128 + (i & 127)] = (_Float16)lWh[i];
  if (lane < 16) {
    float s = 0.f; const float* r = &lWh[lane * 128];
    for (int j = 0; j < 128; ++j) s += r[j] * avec[j];
    s1[i0 + lane] = s;
  } else {
    int m = lane - 16; float s = 0.f; const float* r = &lWh[m * 128];
    for (int j = 0; j < 128; ++j) s += r[j] * avec[128 + j];
    s2[i0 + m] = s;
  }
}

// ---------------- K2b: per-row masked softmax stats (max, den) -------------
__global__ void __launch_bounds__(256)
k2b_stats(const float* __restrict__ s1, const float* __restrict__ s2,
          const int* __restrict__ sec, float* __restrict__ rmax, float* __restrict__ rden) {
  __shared__ float red[256];
  const int i = blockIdx.x, tid = threadIdx.x;
  const float a = s1[i]; const int si = sec[i];
  float mx = -3.0e38f;
  for (int j = tid; j < 2048; j += 256) {
    float v = -9.0e15f;
    if (sec[j] == si) { v = a + s2[j]; v = v > 0.f ? v : 0.2f * v; }
    mx = fmaxf(mx, v);
  }
  red[tid] = mx; __syncthreads();
  for (int s = 128; s > 0; s >>= 1) { if (tid < s) red[tid] = fmaxf(red[tid], red[tid + s]); __syncthreads(); }
  mx = red[0]; __syncthreads();
  float den = 0.f;
  for (int j = tid; j < 2048; j += 256)
    if (sec[j] == si) { float v = a + s2[j]; v = v > 0.f ? v : 0.2f * v; den += __expf(v - mx); }
  red[tid] = den; __syncthreads();
  for (int s = 128; s > 0; s >>= 1) { if (tid < s) red[tid] += red[tid + s]; __syncthreads(); }
  if (tid == 0) { rmax[i] = mx; rden[i] = red[0]; }
}

// --------- K2c: intra = elu(softmax(masked e) @ Wh), streamed WMMA ---------
__global__ void __launch_bounds__(32)
k2c_att(const _Float16* __restrict__ Wh16, const float* __restrict__ s1,
        const float* __restrict__ s2, const float* __restrict__ rmax,
        const float* __restrict__ rden, const int* __restrict__ sec,
        float* __restrict__ intraF, _Float16* __restrict__ intraH) {
  __shared__ float ls2[2048];
  __shared__ int   lsec[2048];
  __shared__ float ls1[16], lmx[16], lin[16];
  __shared__ int   lsi[16];
  const int lane = threadIdx.x, mrow = lane & 15, half = lane >> 4;
  const int i0 = blockIdx.x * 16;
  for (int j = lane; j < 2048; j += 32) { ls2[j] = s2[j]; lsec[j] = sec[j]; }
  if (lane < 16) {
    ls1[lane] = s1[i0 + lane]; lmx[lane] = rmax[i0 + lane];
    lin[lane] = 1.f / rden[i0 + lane]; lsi[lane] = sec[i0 + lane];
  }
  __syncthreads();
  const float s1i = ls1[mrow], mxi = lmx[mrow], invd = lin[mrow];
  const int sii = lsi[mrow];
  v8f acc[8];
  #pragma unroll
  for (int n = 0; n < 8; ++n) {
    v8f z = {};
    acc[n] = z;
  }
  for (int c = 0; c < 64; ++c) {
    v16h a;
    #pragma unroll
    for (int e = 0; e < 16; ++e) {
      int k = c * 32 + (e < 8 ? half * 8 + e : 16 + half * 8 + (e - 8));
      float p = 0.f;
      if (lsec[k] == sii) {
        float v = s1i + ls2[k];
        v = v > 0.f ? v : 0.2f * v;
        p = __expf(v - mxi) * invd;
      }
      a[e] = (_Float16)p;
    }
    #pragma unroll
    for (int nt = 0; nt < 8; ++nt)
      acc[nt] = wmma16(a, loadB(Wh16, 128, lane, c, nt), acc[nt]);
  }
  #pragma unroll
  for (int nt = 0; nt < 8; ++nt) {
    #pragma unroll
    for (int r = 0; r < 8; ++r) {
      int m = r + 8 * half;
      float v = acc[nt][r];
      v = v > 0.f ? v : (__expf(v) - 1.f);
      size_t o = (size_t)(i0 + m) * 128 + nt * 16 + mrow;
      intraF[o] = v; intraH[o] = (_Float16)v;
    }
  }
}

// -------- K3: lg = single-step GRU(intra) (h0=0 -> gh is just bhh) ---------
__global__ void __launch_bounds__(32)
k3_lg(const _Float16* __restrict__ intraH, const _Float16* __restrict__ B3,
      const float* __restrict__ bih, const float* __restrict__ bhh,
      float* __restrict__ lgF, _Float16* __restrict__ fusedX) {
  __shared__ float lG[16 * 384];
  const int lane = threadIdx.x, mrow = lane & 15, half = lane >> 4;
  const int i0 = blockIdx.x * 16;
  v16h af[4];
  #pragma unroll
  for (int c = 0; c < 4; ++c) af[c] = loadA(intraH + (size_t)i0 * 128, 128, mrow, half, c);
  #pragma unroll
  for (int nt = 0; nt < 24; ++nt) {
    v8f acc = {};
    #pragma unroll
    for (int c = 0; c < 4; ++c) acc = wmma16(af[c], loadB(B3, 384, lane, c, nt), acc);
    #pragma unroll
    for (int r = 0; r < 8; ++r) lG[(r + 8 * half) * 384 + nt * 16 + mrow] = acc[r];
  }
  __syncthreads();
  for (int u = 0; u < 64; ++u) {
    int idx = lane * 64 + u, m = idx >> 7, j = idx & 127;
    float rv = sigmoidf_(lG[m * 384 + j] + bih[j] + bhh[j]);
    float zv = sigmoidf_(lG[m * 384 + 128 + j] + bih[128 + j] + bhh[128 + j]);
    float nv = tanhf(lG[m * 384 + 256 + j] + bih[256 + j] + rv * bhh[256 + j]);
    float h = (1.f - zv) * nv;
    lgF[(size_t)(i0 + m) * 128 + j] = h;
    fusedX[(size_t)(i0 + m) * 384 + j] = (_Float16)h;
  }
}

// ------------- K_la: 32-step GRU over windows (H=128 -> E=128) -------------
__global__ void __launch_bounds__(128)
k_la_gru(const float* __restrict__ shortF, const _Float16* __restrict__ Bla,
         const float* __restrict__ bih, const float* __restrict__ bhh,
         const float* __restrict__ aw, const float* __restrict__ ab,
         float* __restrict__ haOut, float* __restrict__ scOut) {
  extern __shared__ char smem[];
  _Float16* lB = (_Float16*)smem;              // 256x384 = 196608 B
  _Float16* lA = (_Float16*)(smem + 196608);   // 16x256
  float* lRZ = (float*)(smem + 204800);        // 16x256
  float* lNX = (float*)(smem + 221184);        // 16x128
  float* lNH = (float*)(smem + 229376);        // 16x128
  float* lHc = (float*)(smem + 237568);        // 16x128
  const int tid = threadIdx.x;
  const int s0 = blockIdx.x * 16;

  { const uint4* s = (const uint4*)Bla; uint4* d = (uint4*)lB;
    for (int i = tid; i < 12288; i += 128) d[i] = s[i]; }
  { uint4 z; z.x = z.y = z.z = z.w = 0u; uint4* d = (uint4*)lA;
    for (int i = tid; i < 512; i += 128) d[i] = z; }
  for (int i = tid; i < 2048; i += 128) lHc[i] = 0.f;
  __syncthreads();

  const int w = __builtin_amdgcn_readfirstlane(tid >> 5);
  const int lane = tid & 31, mrow = lane & 15, half = lane >> 4;

  for (int t = 0; t < 32; ++t) {
    #pragma unroll
    for (int u = 0; u < 16; ++u) {
      int idx = tid * 16 + u, m = idx >> 7, f = idx & 127;
      lA[m * 256 + f] = (_Float16)shortF[((size_t)(s0 + m) * 32 + t) * 128 + f];
    }
    __syncthreads();
    v16h af[8];
    #pragma unroll
    for (int c = 0; c < 8; ++c) af[c] = loadA(lA, 256, mrow, half, c);
    #pragma unroll
    for (int i = 0; i < 4; ++i) {
      int nt = w * 4 + i;
      v8f acc = {};
      #pragma unroll
      for (int c = 0; c < 8; ++c) acc = wmma16(af[c], loadB(lB, 384, lane, c, nt), acc);
      #pragma unroll
      for (int r = 0; r < 8; ++r) lRZ[(r + 8 * half) * 256 + nt * 16 + mrow] = acc[r];
    }
    #pragma unroll
    for (int i = 0; i < 2; ++i) {
      int nt = 16 + w * 2 + i;
      v8f ax = {}, ah = {};
      #pragma unroll
      for (int c = 0; c < 4; ++c) ax = wmma16(af[c], loadB(lB, 384, lane, c, nt), ax);
      #pragma unroll
      for (int c = 4; c < 8; ++c) ah = wmma16(af[c], loadB(lB, 384, lane, c, nt), ah);
      #pragma unroll
      for (int r = 0; r < 8; ++r) {
        lNX[(r + 8 * half) * 128 + (nt - 16) * 16 + mrow] = ax[r];
        lNH[(r + 8 * half) * 128 + (nt - 16) * 16 + mrow] = ah[r];
      }
    }
    __syncthreads();
    #pragma unroll
    for (int u = 0; u < 16; ++u) {
      int idx = tid * 16 + u, m = idx >> 7, j = idx & 127;
      float rv = sigmoidf_(lRZ[m * 256 + j] + bih[j] + bhh[j]);
      float zv = sigmoidf_(lRZ[m * 256 + 128 + j] + bih[128 + j] + bhh[128 + j]);
      float nv = tanhf(lNX[m * 128 + j] + bih[256 + j] + rv * (lNH[m * 128 + j] + bhh[256 + j]));
      float hn = (1.f - zv) * nv + zv * lHc[m * 128 + j];
      lHc[m * 128 + j] = hn;
      lA[m * 256 + 128 + j] = (_Float16)hn;
      haOut[((size_t)(s0 + m) * 32 + t) * 128 + j] = hn;
    }
    __syncthreads();
    if (tid < 16) {
      float s = ab[0];
      const float* hp = &lHc[tid * 128];
      for (int j = 0; j < 128; ++j) s += hp[j] * aw[j];
      scOut[(size_t)(s0 + tid) * 32 + t] = s;
    }
  }
}

// ------------------- K3b: la attention pooling over 32 steps ---------------
__global__ void __launch_bounds__(128)
k3b_lapool(const float* __restrict__ ha, const float* __restrict__ sc,
           _Float16* __restrict__ fusedX) {
  __shared__ float w[32];
  const int s = blockIdx.x, tid = threadIdx.x;
  if (tid == 0) {
    float mx = -3.0e38f;
    for (int t = 0; t < 32; ++t) mx = fmaxf(mx, sc[s * 32 + t]);
    float den = 0.f;
    for (int t = 0; t < 32; ++t) { float e = __expf(sc[s * 32 + t] - mx); w[t] = e; den += e; }
    float inv = 1.f / den;
    for (int t = 0; t < 32; ++t) w[t] *= inv;
  }
  __syncthreads();
  float acc = 0.f;
  const float* hp = ha + (size_t)s * 32 * 128 + tid;
  for (int t = 0; t < 32; ++t) acc += w[t] * hp[t * 128];
  fusedX[(size_t)s * 384 + 128 + tid] = (_Float16)acc;
}

// ----------------------- K4: sector mean (segment mean) --------------------
__global__ void __launch_bounds__(128)
k4_secmean(const float* __restrict__ lgF, const int* __restrict__ sec,
           float* __restrict__ secH) {
  __shared__ float acc[16 * 128];
  __shared__ float cnt[16];
  const int j = threadIdx.x;
  for (int si = 0; si < 16; ++si) acc[si * 128 + j] = 0.f;
  if (j < 16) cnt[j] = 0.f;
  __syncthreads();
  for (int r = 0; r < 2048; ++r) {
    int si = sec[r];
    acc[si * 128 + j] += lgF[(size_t)r * 128 + j];
    if (j == 0) cnt[si] += 1.f;
  }
  __syncthreads();
  for (int si = 0; si < 16; ++si)
    secH[si * 128 + j] = acc[si * 128 + j] / fmaxf(cnt[si], 1.f);
}

// ----------------------- K5: inter-sector GAT (16 nodes) -------------------
__global__ void __launch_bounds__(128)
k5_intergat(const float* __restrict__ secH, const float* __restrict__ W,
            const float* __restrict__ avec, const int* __restrict__ adj,
            float* __restrict__ secOut) {
  __shared__ float lWh[16 * 128];
  __shared__ float ls1[16], ls2[16];
  __shared__ float att[16 * 16];
  const int j = threadIdx.x;
  for (int i = 0; i < 16; ++i) {
    float s = 0.f;
    for (int k = 0; k < 128; ++k) s += secH[i * 128 + k] * W[k * 128 + j];
    lWh[i * 128 + j] = s;
  }
  __syncthreads();
  if (j < 16) {
    float a0 = 0.f, a1 = 0.f;
    for (int k = 0; k < 128; ++k) { a0 += lWh[j * 128 + k] * avec[k]; a1 += lWh[j * 128 + k] * avec[128 + k]; }
    ls1[j] = a0; ls2[j] = a1;
  }
  __syncthreads();
  if (j < 16) {
    float e[16]; float mx = -3.4e38f;
    for (int k = 0; k < 16; ++k) {
      float v = -9.0e15f;
      if (adj[j * 16 + k] > 0) { v = ls1[j] + ls2[k]; v = v > 0.f ? v : 0.2f * v; }
      e[k] = v; mx = fmaxf(mx, v);
    }
    float den = 0.f;
    for (int k = 0; k < 16; ++k) { e[k] = __expf(e[k] - mx); den += e[k]; }
    float inv = 1.f / den;
    for (int k = 0; k < 16; ++k) att[j * 16 + k] = e[k] * inv;
  }
  __syncthreads();
  for (int i = 0; i < 16; ++i) {
    float s = 0.f;
    for (int k = 0; k < 16; ++k) s += att[i * 16 + k] * lWh[k * 128 + j];
    s = s > 0.f ? s : (__expf(s) - 1.f);
    secOut[i * 128 + j] = s;
  }
}

// ----------------- K5b: gather sector embedding per stock ------------------
__global__ void __launch_bounds__(256)
k5b_gather(const float* __restrict__ secOut, const int* __restrict__ sec,
           _Float16* __restrict__ fusedX) {
  int idx = blockIdx.x * 256 + threadIdx.x;
  if (idx >= 2048 * 128) return;
  int s = idx >> 7, j = idx & 127;
  fusedX[(size_t)s * 384 + 256 + j] = (_Float16)secOut[sec[s] * 128 + j];
}

// ---------------- K6: fusion GEMM + return/movement heads ------------------
__global__ void __launch_bounds__(32)
k6_fusion(const _Float16* __restrict__ fusedX, const _Float16* __restrict__ Bf,
          const float* __restrict__ fb, const float* __restrict__ rw,
          const float* __restrict__ rb, const float* __restrict__ mw,
          const float* __restrict__ mb, float* __restrict__ out) {
  __shared__ float lF[16 * 128];
  const int lane = threadIdx.x, mrow = lane & 15, half = lane >> 4;
  const int i0 = blockIdx.x * 16;
  v16h af[12];
  #pragma unroll
  for (int c = 0; c < 12; ++c) af[c] = loadA(fusedX + (size_t)i0 * 384, 384, mrow, half, c);
  #pragma unroll
  for (int nt = 0; nt < 8; ++nt) {
    v8f acc = {};
    #pragma unroll
    for (int c = 0; c < 12; ++c) acc = wmma16(af[c], loadB(Bf, 128, lane, c, nt), acc);
    #pragma unroll
    for (int r = 0; r < 8; ++r) lF[(r + 8 * half) * 128 + nt * 16 + mrow] = acc[r];
  }
  __syncthreads();
  if (lane < 16) {
    float rr = rb[0], mm = mb[0];
    const float* row = &lF[lane * 128];
    for (int j = 0; j < 128; ++j) {
      float v = row[j] + fb[j];
      rr += v * rw[j]; mm += v * mw[j];
    }
    out[i0 + lane] = rr;
    out[2048 + i0 + lane] = sigmoidf_(mm);
  }
}

// ---------------------------------------------------------------------------
extern "C" void kernel_launch(void* const* d_in, const int* in_sizes, int n_in,
                              void* d_out, int out_size, void* d_ws, size_t ws_size,
                              hipStream_t stream) {
  (void)in_sizes; (void)n_in; (void)out_size; (void)ws_size;
  const float* xfeat = (const float*)d_in[0];
  const int*   sec   = (const int*)d_in[1];
  const int*   adj   = (const int*)d_in[2];
  const float* g1Wih = (const float*)d_in[3];
  const float* g1Whh = (const float*)d_in[4];
  const float* g1bih = (const float*)d_in[5];
  const float* g1bhh = (const float*)d_in[6];
  const float* a1w   = (const float*)d_in[7];
  const float* a1b   = (const float*)d_in[8];
  const float* giW   = (const float*)d_in[9];
  const float* giA   = (const float*)d_in[10];
  const float* ggWih = (const float*)d_in[11];
  const float* ggbih = (const float*)d_in[13];
  const float* ggbhh = (const float*)d_in[14];
  const float* gaWih = (const float*)d_in[17];
  const float* gaWhh = (const float*)d_in[18];
  const float* gabih = (const float*)d_in[19];
  const float* gabhh = (const float*)d_in[20];
  const float* aaw   = (const float*)d_in[21];
  const float* aab   = (const float*)d_in[22];
  const float* geW   = (const float*)d_in[23];
  const float* geA   = (const float*)d_in[24];
  const float* fw    = (const float*)d_in[25];
  const float* fb    = (const float*)d_in[26];
  const float* rw    = (const float*)d_in[27];
  const float* rb    = (const float*)d_in[28];
  const float* mw    = (const float*)d_in[29];
  const float* mb    = (const float*)d_in[30];
  float* out = (float*)d_out;

  char* ws = (char*)d_ws;
  float*    wsShort  = (float*)(ws + 0);               // 65536x128 f32
  float*    wsHa     = (float*)(ws + 33554432);        // 2048x32x128 f32
  _Float16* wsB1     = (_Float16*)(ws + 67108864);     // 160x384 f16
  _Float16* wsWintra = (_Float16*)(ws + 67231744);     // 128x128 f16
  _Float16* wsWh16   = (_Float16*)(ws + 67264512);     // 2048x128 f16
  float*    wsS1     = (float*)(ws + 67788800);        // 2048
  float*    wsS2     = (float*)(ws + 67796992);        // 2048
  float*    wsRmax   = (float*)(ws + 67805184);        // 2048
  float*    wsRden   = (float*)(ws + 67813376);        // 2048
  float*    wsIntraF = (float*)(ws + 67821568);        // 2048x128 f32
  _Float16* wsIntraH = (_Float16*)(ws + 68870144);     // 2048x128 f16
  _Float16* wsB3     = (_Float16*)(ws + 69394432);     // 128x384 f16
  float*    wsLg     = (float*)(ws + 69492736);        // 2048x128 f32
  _Float16* wsBla    = (_Float16*)(ws + 70541312);     // 256x384 f16
  float*    wsSc     = (float*)(ws + 70737920);        // 2048x32 f32
  _Float16* wsFx     = (_Float16*)(ws + 71000064);     // 2048x384 f16
  _Float16* wsBfus   = (_Float16*)(ws + 72572928);     // 384x128 f16
  float*    wsSecH   = (float*)(ws + 72671232);        // 16x128
  float*    wsSecO   = (float*)(ws + 72679424);        // 16x128

  // ---- weight prep: f16 casts/transposes into WMMA-B layouts ----
  k_tcast<<<(16 * 384 + 255) / 256, 256, 0, stream>>>(wsB1, g1Wih, 16, 384);
  k_zeroh<<<(16 * 384 + 255) / 256, 256, 0, stream>>>(wsB1 + 16 * 384, 16 * 384);
  k_tcast<<<(128 * 384 + 255) / 256, 256, 0, stream>>>(wsB1 + 32 * 384, g1Whh, 128, 384);
  k_cast16<<<(128 * 128 + 255) / 256, 256, 0, stream>>>(giW, wsWintra, 128 * 128);
  k_tcast<<<(128 * 384 + 255) / 256, 256, 0, stream>>>(wsB3, ggWih, 128, 384);
  k_tcast<<<(128 * 384 + 255) / 256, 256, 0, stream>>>(wsBla, gaWih, 128, 384);
  k_tcast<<<(128 * 384 + 255) / 256, 256, 0, stream>>>(wsBla + 128 * 384, gaWhh, 128, 384);
  k_cast16<<<(384 * 128 + 255) / 256, 256, 0, stream>>>(fw, wsBfus, 384 * 128);

  // ---- stage 1: 65536 window GRUs + temporal attention ----
  k1_window_gru<<<4096, 128, 202048, stream>>>(xfeat, wsB1, g1bih, g1bhh, a1w, a1b, wsShort);

  // ---- stage 2: intra-sector GAT ----
  k2a_wh<<<128, 32, 0, stream>>>(wsShort, wsWintra, giA, wsWh16, wsS1, wsS2);
  k2b_stats<<<2048, 256, 0, stream>>>(wsS1, wsS2, sec, wsRmax, wsRden);
  k2c_att<<<128, 32, 0, stream>>>(wsWh16, wsS1, wsS2, wsRmax, wsRden, sec, wsIntraF, wsIntraH);

  // ---- stage 3: lg (single-step GRU) and la (32-step GRU + attention) ----
  k3_lg<<<128, 32, 0, stream>>>(wsIntraH, wsB3, ggbih, ggbhh, wsLg, wsFx);
  k_la_gru<<<128, 128, 245760, stream>>>(wsShort, wsBla, gabih, gabhh, aaw, aab, wsHa, wsSc);
  k3b_lapool<<<2048, 128, 0, stream>>>(wsHa, wsSc, wsFx);

  // ---- stage 4: sector pooling + inter-sector GAT + gather ----
  k4_secmean<<<1, 128, 0, stream>>>(wsLg, sec, wsSecH);
  k5_intergat<<<1, 128, 0, stream>>>(wsSecH, geW, geA, adj, wsSecO);
  k5b_gather<<<1024, 256, 0, stream>>>(wsSecO, sec, wsFx);

  // ---- stage 5: fusion + heads ----
  k6_fusion<<<128, 32, 0, stream>>>(wsFx, wsBfus, fb, rw, rb, mw, mb, out);
}
